// NeuralVirtualizationLayer_44135083933809
// MI455X (gfx1250) — matrix operand order repaired and checked by reading, not compiled, not run on hardware
//
#include <hip/hip_runtime.h>
#include <hip/hip_bf16.h>
#include <math.h>

// ---------------------------------------------------------------------------
// Problem constants (from reference): B=32 A=256 IN=512 R=256 D=256 M=2048
// H=1024 S=128.  mmu_out width = 5*D+4 = 1284.
// ---------------------------------------------------------------------------
#define CB   32
#define CA   256
#define CIN  512
#define CR   256
#define CD   256
#define CM   2048
#define CH   1024
#define CS   128
#define CMW  1284          // mmu_out columns

typedef _Float16 half_t;
typedef __attribute__((ext_vector_type(16))) _Float16 v16h;
typedef __attribute__((ext_vector_type(8)))  float    v8f;

union HF { unsigned u[8]; v16h h; };

// ---------------------------------------------------------------------------
// f16-in / f16-WMMA / f32-out GEMM (all operands pre-converted to f16):
//   C[m,n] = sum_k Aeff[m,k] * Beff[k,n] (+ bias[n]);  optional f16 mirror C16
//   Aeff[m,k] = TA ? A[k*lda+m] : A[m*lda+k]
//   Beff[k,n] = TB ? B[n*ldb+k] : B[k*ldb+n]
// Block = 256 threads = 8 waves arranged 4(M) x 2(N); wave tile 32x64
// (2 A-frags x 4 B-frags -> 8 v_wmma per K-step, B reused across A).
// Contiguous paths load packed f16 pairs as u32 (merge into b128 clauses)
// and prefetch the next K-chunk (global_prefetch_b8).
// Requires M % 128 == 0 and K % 32 == 0.  nbase offsets the N tiling so an
// unmasked bulk launch can be paired with a masked edge launch (NCHK=true).
// EXEC stays all-ones for every WMMA.
// ---------------------------------------------------------------------------
template <bool TA, bool TB, bool NCHK>
__global__ void __launch_bounds__(256)
wmma_gemm16(const half_t* __restrict__ Ag, const half_t* __restrict__ Bg,
            float* __restrict__ Cg, const float* __restrict__ bias,
            half_t* __restrict__ C16g, int nbase,
            int M, int N, int K, int lda, int ldb, int ldc,
            long long sA, long long sB, long long sC)
{
    const int wave = threadIdx.x >> 5;
    const int lane = threadIdx.x & 31;

    const half_t* A = Ag + (long long)blockIdx.z * sA;
    const half_t* B = Bg + (long long)blockIdx.z * sB;
    float*        C = Cg + (long long)blockIdx.z * sC;
    half_t*       C16 = C16g ? C16g + (long long)blockIdx.z * sC : (half_t*)0;

    const int m0 = blockIdx.y * 128 + (wave & 3) * 32;           // 2 sub-tiles of 16
    const int n0 = nbase + blockIdx.x * 128 + (wave >> 2) * 64;  // 4 sub-tiles of 16

    v8f acc[2][4];
#pragma unroll
    for (int h = 0; h < 2; ++h)
#pragma unroll
        for (int t = 0; t < 4; ++t)
#pragma unroll
            for (int r = 0; r < 8; ++r) acc[h][t][r] = 0.0f;

    // A-fragment lane mapping (16-bit A 16x32, ISA 7.12.2)
    const int am = lane & 15;                  // M within tile
    const int ag = lane >> 4;                  // K group select
    // B-fragment lane mapping (16-bit B 32x16)
    const int bn = lane & 15;                  // N within tile
    const int bh = lane >> 4;                  // K group (0 -> K0-7,16-23; 1 -> K8-15,24-31)

    // Pre-resolved (clamped) B column index + validity per N sub-tile.
    int  ncol[4];
    bool nok[4];
#pragma unroll
    for (int t = 0; t < 4; ++t) {
        const int n = n0 + t * 16 + bn;
        nok[t]  = (!NCHK) || (n < N);
        ncol[t] = NCHK ? (nok[t] ? n : 0) : n;   // index 0 is always in-bounds
    }

    for (int kb = 0; kb < K; kb += 32) {
        HF af[2];
#pragma unroll
        for (int h = 0; h < 2; ++h) {
            const int arow = m0 + h * 16 + am;
            if (TA) {
#pragma unroll
                for (int j = 0; j < 4; ++j) {
                    const int k0 = kb + ag * 8 + 2 * j;
                    const int k1 = k0 + 16;
                    af[h].h[2 * j]         = A[(size_t)k0 * lda + arow];
                    af[h].h[2 * j + 1]     = A[(size_t)(k0 + 1) * lda + arow];
                    af[h].h[8 + 2 * j]     = A[(size_t)k1 * lda + arow];
                    af[h].h[8 + 2 * j + 1] = A[(size_t)(k1 + 1) * lda + arow];
                }
            } else {
                const half_t* ap = A + (size_t)arow * lda + kb + ag * 8;
#pragma unroll
                for (int j = 0; j < 4; ++j) {
                    af[h].u[j]     = *(const unsigned*)(ap + 2 * j);
                    af[h].u[4 + j] = *(const unsigned*)(ap + 16 + 2 * j);
                }
                __builtin_prefetch((const void*)(ap + 32), 0, 1);  // next K chunk
            }
        }
#pragma unroll
        for (int t = 0; t < 4; ++t) {
            HF bf;
            if (TB) {
                const half_t* bp = B + (size_t)ncol[t] * ldb + kb + bh * 8;
#pragma unroll
                for (int j = 0; j < 8; ++j) bf.u[j] = *(const unsigned*)(bp + 2 * j + (j >= 4 ? 8 : 0));
                __builtin_prefetch((const void*)(bp + 32), 0, 1);  // next K chunk
            } else {
#pragma unroll
                for (int j = 0; j < 8; ++j) {
                    const int k = kb + bh * 8 + 2 * j + (j >= 4 ? 8 : 0);
                    bf.h[2 * j]     = B[(size_t)k * ldb + ncol[t]];
                    bf.h[2 * j + 1] = B[(size_t)(k + 1) * ldb + ncol[t]];
                }
            }
            if (NCHK) {
#pragma unroll
                for (int j = 0; j < 8; ++j) bf.u[j] = nok[t] ? bf.u[j] : 0u;
            }
            acc[0][t] = __builtin_amdgcn_wmma_f32_16x16x32_f16(
                false, af[0].h, false, bf.h, (short)0, acc[0][t], false, false);
            acc[1][t] = __builtin_amdgcn_wmma_f32_16x16x32_f16(
                false, af[1].h, false, bf.h, (short)0, acc[1][t], false, false);
        }
    }

    // C/D layout: VGPR r -> row (+8 for upper lane half), lane&15 -> col
    const int cn = lane & 15;
    const int cm = (lane >> 4) * 8;
#pragma unroll
    for (int t = 0; t < 4; ++t) {
        const int col = n0 + t * 16 + cn;
        if (NCHK && col >= N) continue;
        const float bv = bias ? bias[col] : 0.0f;
#pragma unroll
        for (int h = 0; h < 2; ++h) {
#pragma unroll
            for (int r = 0; r < 8; ++r) {
                const int row = m0 + h * 16 + cm + r;
                const float v = acc[h][t][r] + bv;
                C[(size_t)row * ldc + col] = v;
                if (C16) C16[(size_t)row * ldc + col] = (half_t)v;
            }
        }
    }
}

// ---------------------------------------------------------------------------
// Vectorized f32 -> f16 conversion (n must be a multiple of 4; all call
// sites satisfy this).
// ---------------------------------------------------------------------------
__global__ void f32_to_f16_v4(const float* __restrict__ in, half_t* __restrict__ out,
                              long long n)
{
    const long long i = ((long long)blockIdx.x * blockDim.x + threadIdx.x) * 4;
    if (i >= n) return;
    const float4 v = *(const float4*)(in + i);
    union { half_t h[4]; uint2 u; } r;
    r.h[0] = (half_t)v.x; r.h[1] = (half_t)v.y;
    r.h[2] = (half_t)v.z; r.h[3] = (half_t)v.w;
    *(uint2*)(out + i) = r.u;
}

// ---------------------------------------------------------------------------
// Row softmax (in place) + optional f16 mirror:
//   p = softmax(p * scale) * (gate ? sigmoid(gate[row]) : 1)
// ---------------------------------------------------------------------------
__global__ void __launch_bounds__(256)
row_softmax(float* __restrict__ data, half_t* __restrict__ mirror, int L, float scale,
            const float* __restrict__ gate, int gstride)
{
    const int row = blockIdx.x;
    float*  p   = data + (size_t)row * L;
    half_t* m16 = mirror ? mirror + (size_t)row * L : (half_t*)0;
    __shared__ float red[256];
    const int t = threadIdx.x;

    float mx = -3.4e38f;
    for (int i = t; i < L; i += 256) mx = fmaxf(mx, p[i] * scale);
    red[t] = mx; __syncthreads();
    for (int s = 128; s > 0; s >>= 1) { if (t < s) red[t] = fmaxf(red[t], red[t + s]); __syncthreads(); }
    mx = red[0]; __syncthreads();

    float sum = 0.0f;
    for (int i = t; i < L; i += 256) { float e = expf(p[i] * scale - mx); p[i] = e; sum += e; }
    red[t] = sum; __syncthreads();
    for (int s = 128; s > 0; s >>= 1) { if (t < s) red[t] += red[t + s]; __syncthreads(); }

    float g = 1.0f;
    if (gate) g = 1.0f / (1.0f + expf(-gate[(size_t)row * gstride]));
    const float inv = g / red[0];
    for (int i = t; i < L; i += 256) {
        const float v = p[i] * inv;
        p[i] = v;
        if (m16) m16[i] = (half_t)v;
    }
}

// gru_in = concat(x broadcast over A, read_data) -> f16 directly
__global__ void build_gru_in(const float* __restrict__ x, const float* __restrict__ rd,
                             half_t* __restrict__ gin, int total)
{
    int idx = blockIdx.x * blockDim.x + threadIdx.x;
    if (idx >= total) return;
    const int row = idx / (CIN + CD);
    const int col = idx - row * (CIN + CD);
    const int b   = row >> 8;                    // row / A
    const float v = (col < CIN) ? x[b * CIN + col]
                                : rd[(size_t)row * CD + (col - CIN)];
    gin[idx] = (half_t)v;
}

// GRU gates -> new_regs (f32 into d_out) + f16 mirror
__global__ void gru_gates(const float* __restrict__ gi, const float* __restrict__ gh,
                          const float* __restrict__ curr, float* __restrict__ nr,
                          half_t* __restrict__ nr16, int total)
{
    int idx = blockIdx.x * blockDim.x + threadIdx.x;
    if (idx >= total) return;
    const int row = idx >> 8;                    // / R
    const int j   = idx & 255;
    const float* gir = gi + (size_t)row * (3 * CR);
    const float* ghr = gh + (size_t)row * (3 * CR);
    const float r = 1.0f / (1.0f + expf(-(gir[j] + ghr[j])));
    const float z = 1.0f / (1.0f + expf(-(gir[CR + j] + ghr[CR + j])));
    const float n = tanhf(gir[2 * CR + j] + r * ghr[2 * CR + j]);
    const float v = (1.0f - z) * n + z * curr[idx];
    nr[idx]   = v;
    nr16[idx] = (half_t)v;
}

// total_w[b,m] = clip(sum_a gated_wattn[b,a,m], 0, 1)
__global__ void colsum_clip(const float* __restrict__ watt, float* __restrict__ tw)
{
    const int m = blockIdx.x * 256 + threadIdx.x;
    const int b = blockIdx.y;
    const float* base = watt + (size_t)b * CA * CM + m;
    float s = 0.0f;
    for (int a = 0; a < CA; ++a) s += base[(size_t)a * CM];
    tw[b * CM + m] = fminf(fmaxf(s, 0.0f), 1.0f);
}

// Per-batch: read_gate = softmax3(ssd_gate)[0]; agg_key; read_mask
__global__ void __launch_bounds__(256)
ssd_reduce(const float* __restrict__ mmu, float* __restrict__ aggk, float* __restrict__ rmask)
{
    const int b = blockIdx.x, t = threadIdx.x;
    __shared__ float s_g[256];
    __shared__ float red[256];
    {
        const size_t base = ((size_t)(b * CA + t)) * CMW + (5 * CD + 1);
        const float l0 = mmu[base], l1 = mmu[base + 1], l2 = mmu[base + 2];
        const float mx = fmaxf(l0, fmaxf(l1, l2));
        const float e0 = expf(l0 - mx), e1 = expf(l1 - mx), e2 = expf(l2 - mx);
        s_g[t] = e0 / (e0 + e1 + e2);
    }
    __syncthreads();
    red[t] = s_g[t]; __syncthreads();
    for (int s = 128; s > 0; s >>= 1) { if (t < s) red[t] += red[t + s]; __syncthreads(); }
    const float rs = red[0];
    __syncthreads();

    float acc = 0.0f;
    for (int a = 0; a < CA; ++a)
        acc += mmu[((size_t)(b * CA + a)) * CMW + (3 * CD + 1) + t] * s_g[a];
    aggk[b * CD + t] = acc / (rs + 1e-6f);
    if (t == 0) {
        const float avg = rs * (1.0f / CA);
        rmask[b] = 1.0f / (1.0f + expf(-(avg - 0.5f) * 10.0f));
    }
}

// Per-batch: ssd attention over S=128 keys, blend into slot0
__global__ void __launch_bounds__(256)
ssd_slot0(const float* __restrict__ aggk, const float* __restrict__ keys,
          const float* __restrict__ vals, const float* __restrict__ rmask,
          const float* __restrict__ mem, float* __restrict__ slot0)
{
    const int b = blockIdx.x, t = threadIdx.x;
    __shared__ float s_p[CS];
    __shared__ float s_inv;
    if (t < CS) {
        const float* kr = keys + (size_t)t * CD;
        const float* ak = aggk + b * CD;
        float dot = 0.0f;
        for (int d = 0; d < CD; ++d) dot += ak[d] * kr[d];
        s_p[t] = dot * 0.0625f;                  // 1/sqrt(D)
    }
    __syncthreads();
    if (t == 0) {
        float mx = -3.4e38f;
        for (int s = 0; s < CS; ++s) mx = fmaxf(mx, s_p[s]);
        float sum = 0.0f;
        for (int s = 0; s < CS; ++s) { float e = expf(s_p[s] - mx); s_p[s] = e; sum += e; }
        s_inv = 1.0f / sum;
    }
    __syncthreads();
    const float inv = s_inv;
    float rv = 0.0f;
    for (int s = 0; s < CS; ++s) rv += s_p[s] * inv * vals[(size_t)s * CD + t];
    const float m = rmask[b];
    slot0[b * CD + t] = (1.0f - m) * mem[(size_t)b * CM * CD + t] + m * rv;
}

// new_mem = base*(1-total_w) + combined ; base row 0 is the blended slot0
__global__ void mem_final(const float* __restrict__ mem, const float* __restrict__ slot0,
                          const float* __restrict__ tw, const float* __restrict__ comb,
                          float* __restrict__ out, int total)
{
    int idx = blockIdx.x * blockDim.x + threadIdx.x;
    if (idx >= total) return;
    const int d  = idx & 255;
    const int bm = idx >> 8;
    const int m  = bm & (CM - 1);
    const int b  = bm >> 11;
    const float t    = tw[bm];
    const float base = (m == 0) ? slot0[b * CD + d] : mem[idx];
    out[idx] = base * (1.0f - t) + comb[idx];
}

// Per-batch aggregation attention + LayerNorm
__global__ void __launch_bounds__(256)
agg_ln(const float* __restrict__ proj, const float* __restrict__ q,
       const float* __restrict__ g, const float* __restrict__ beta,
       float* __restrict__ out)
{
    const int b = blockIdx.x, t = threadIdx.x;
    __shared__ float s_p[256];
    __shared__ float red[256];
    __shared__ float s_agg[CH];

    const float* pr = proj + ((size_t)(b * CA + t)) * CH;
    float dot = 0.0f;
    for (int h = 0; h < CH; ++h) dot += q[h] * pr[h];
    s_p[t] = dot * 0.03125f;                     // 1/sqrt(H)
    __syncthreads();

    red[t] = s_p[t]; __syncthreads();
    for (int s = 128; s > 0; s >>= 1) { if (t < s) red[t] = fmaxf(red[t], red[t + s]); __syncthreads(); }
    const float mx = red[0]; __syncthreads();
    const float e = expf(s_p[t] - mx);
    s_p[t] = e; red[t] = e; __syncthreads();
    for (int s = 128; s > 0; s >>= 1) { if (t < s) red[t] += red[t + s]; __syncthreads(); }
    s_p[t] *= 1.0f / red[0];
    __syncthreads();

    for (int i = 0; i < 4; ++i) {
        const int h = t + 256 * i;
        float a = 0.0f;
        for (int aa = 0; aa < CA; ++aa)
            a += s_p[aa] * proj[((size_t)(b * CA + aa)) * CH + h];
        s_agg[h] = a;
    }
    __syncthreads();

    float ls = 0.0f;
    for (int i = 0; i < 4; ++i) ls += s_agg[t + 256 * i];
    red[t] = ls; __syncthreads();
    for (int s = 128; s > 0; s >>= 1) { if (t < s) red[t] += red[t + s]; __syncthreads(); }
    const float mu = red[0] * (1.0f / CH); __syncthreads();
    float lq = 0.0f;
    for (int i = 0; i < 4; ++i) { const float d = s_agg[t + 256 * i] - mu; lq += d * d; }
    red[t] = lq; __syncthreads();
    for (int s = 128; s > 0; s >>= 1) { if (t < s) red[t] += red[t + s]; __syncthreads(); }
    const float rstd = rsqrtf(red[0] * (1.0f / CH) + 1e-5f);
    for (int i = 0; i < 4; ++i) {
        const int h = t + 256 * i;
        out[b * CH + h] = (s_agg[h] - mu) * rstd * g[h] + beta[h];
    }
}

// ---------------------------------------------------------------------------
// Workspace layout.  f32 region first (element offsets), then f16 region.
// ---------------------------------------------------------------------------
static const long long SZ_MMU  = (long long)CB * CA * CMW;        // 10,518,528
static const long long SZ_AM   = (long long)CB * CA * CM;         // 16,777,216
static const long long OFF_MMU  = 0;
static const long long OFF_ATTN = OFF_MMU + SZ_MMU;               // attn logits, later combined
static const long long OFF_WATT = OFF_ATTN + SZ_AM;
static const long long OFF_READ = OFF_WATT + SZ_AM;               // B*A*D
static const long long OFF_GI   = OFF_READ + (long long)CB * CA * CD;
static const long long OFF_GH   = OFF_GI + (long long)CB * CA * 3 * CR;
static const long long OFF_OUTA = OFF_GH + (long long)CB * CA * 3 * CR;
static const long long OFF_PROJ = OFF_OUTA + (long long)CB * CA * CIN;
static const long long OFF_TW   = OFF_PROJ + (long long)CB * CA * CH;
static const long long OFF_AGGK = OFF_TW + (long long)CB * CM;
static const long long OFF_MASK = OFF_AGGK + (long long)CB * CD;
static const long long OFF_SLOT = OFF_MASK + CB;
static const long long OFF_F16  = OFF_SLOT + (long long)CB * CD;  // f16 region starts here (f32 units)

// f16 element offsets inside the f16 region
static const long long H_HID   = 0;                               // B*A*R
static const long long H_MEM   = H_HID  + (long long)CB * CA * CR;
static const long long H_MMUW  = H_MEM  + (long long)CB * CM * CD;
static const long long H_WIH   = H_MMUW + (long long)CMW * CR;
static const long long H_WHH   = H_WIH  + (long long)3 * CR * (CIN + CD);
static const long long H_OUTW  = H_WHH  + (long long)3 * CR * CR;
static const long long H_PROJW = H_OUTW + (long long)CIN * CR;
static const long long H_MMU   = H_PROJW + (long long)CH * CIN;   // mmu_out mirror
static const long long H_ATTNP = H_MMU  + SZ_MMU;                 // attn probs mirror
static const long long H_WATTG = H_ATTNP + SZ_AM;                 // gated w_attn mirror
static const long long H_GIN   = H_WATTG + SZ_AM;                 // gru_in
static const long long H_NR    = H_GIN  + (long long)CB * CA * (CIN + CD);
static const long long H_OUTA  = H_NR   + (long long)CB * CA * CR;

extern "C" void kernel_launch(void* const* d_in, const int* in_sizes, int n_in,
                              void* d_out, int out_size, void* d_ws, size_t ws_size,
                              hipStream_t stream)
{
    const float* x        = (const float*)d_in[0];
    const float* hidden   = (const float*)d_in[1];
    const float* memstate = (const float*)d_in[2];
    const float* mmu_w    = (const float*)d_in[3];
    const float* mmu_b    = (const float*)d_in[4];
    const float* gru_wih  = (const float*)d_in[5];
    const float* gru_whh  = (const float*)d_in[6];
    const float* gru_bih  = (const float*)d_in[7];
    const float* gru_bhh  = (const float*)d_in[8];
    const float* out_w    = (const float*)d_in[9];
    const float* out_b    = (const float*)d_in[10];
    const float* agg_q    = (const float*)d_in[11];
    const float* agg_pw   = (const float*)d_in[12];
    const float* agg_pb   = (const float*)d_in[13];
    const float* ln_g     = (const float*)d_in[14];
    const float* ln_b     = (const float*)d_in[15];
    const float* ssd_keys = (const float*)d_in[16];
    const float* ssd_vals = (const float*)d_in[17];

    float* out    = (float*)d_out;
    float* o_agg  = out;                                   // 32*1024
    float* o_hid  = out + (long long)CB * CH;              // 32*256*256
    float* o_mem  = o_hid + (long long)CB * CA * CR;       // 32*2048*256

    float*  ws = (float*)d_ws;
    half_t* hb = (half_t*)(ws + OFF_F16);
    const dim3 blk(256);

#define CVT(src, dstoff, n) \
    f32_to_f16_v4<<<dim3((unsigned)(((n) / 4 + 255) / 256)), blk, 0, stream>>>(src, hb + (dstoff), (long long)(n))

    // 0) one-time f16 conversions of all GEMM operands
    CVT(hidden,   H_HID,   (long long)CB * CA * CR);
    CVT(memstate, H_MEM,   (long long)CB * CM * CD);
    CVT(mmu_w,    H_MMUW,  (long long)CMW * CR);
    CVT(gru_wih,  H_WIH,   (long long)3 * CR * (CIN + CD));
    CVT(gru_whh,  H_WHH,   (long long)3 * CR * CR);
    CVT(out_w,    H_OUTW,  (long long)CIN * CR);
    CVT(agg_pw,   H_PROJW, (long long)CH * CIN);
#undef CVT

    // 1a) mmu_out bulk: columns 0..1279, no masking   (8192 x 1280 x 256)
    wmma_gemm16<false, true, false><<<dim3(10, 64, 1), blk, 0, stream>>>(
        hb + H_HID, hb + H_MMUW, ws + OFF_MMU, mmu_b, hb + H_MMU, 0,
        CB * CA, CMW, CR, CR, CR, CMW, 0, 0, 0);

    // 1b) mmu_out edge: columns 1280..1283, masked    (8192 x 4 x 256)
    wmma_gemm16<false, true, true><<<dim3(1, 64, 1), blk, 0, stream>>>(
        hb + H_HID, hb + H_MMUW, ws + OFF_MMU, mmu_b, hb + H_MMU, 1280,
        CB * CA, CMW, CR, CR, CR, CMW, 0, 0, 0);

    // 2) attn logits = query @ mem^T              (per-b 256 x 2048 x 256)
    wmma_gemm16<false, true, false><<<dim3(16, 2, CB), blk, 0, stream>>>(
        hb + H_MMU, hb + H_MEM, ws + OFF_ATTN, nullptr, nullptr, 0,
        CA, CM, CD, CMW, CD, CM,
        (long long)CA * CMW, (long long)CM * CD, (long long)CA * CM);

    // 3) softmax over M (+ f16 mirror of probabilities)
    row_softmax<<<dim3(CB * CA), blk, 0, stream>>>(ws + OFF_ATTN, hb + H_ATTNP,
                                                   CM, 0.0625f, nullptr, 0);

    // 4) read_data = attn @ mem                   (per-b 256 x 256 x 2048)
    wmma_gemm16<false, false, false><<<dim3(2, 2, CB), blk, 0, stream>>>(
        hb + H_ATTNP, hb + H_MEM, ws + OFF_READ, nullptr, nullptr, 0,
        CA, CD, CM, CM, CD, CD,
        (long long)CA * CM, (long long)CM * CD, (long long)CA * CD);

    // 5) gru_in = [x | read_data] (f16)
    {
        const int total = CB * CA * (CIN + CD);
        build_gru_in<<<dim3((total + 255) / 256), blk, 0, stream>>>(x, ws + OFF_READ,
                                                                    hb + H_GIN, total);
    }

    // 6) gi = gru_in @ Wih^T + bih                (8192 x 768 x 768)
    wmma_gemm16<false, true, false><<<dim3(6, 64, 1), blk, 0, stream>>>(
        hb + H_GIN, hb + H_WIH, ws + OFF_GI, gru_bih, nullptr, 0,
        CB * CA, 3 * CR, CIN + CD, CIN + CD, CIN + CD, 3 * CR, 0, 0, 0);

    // 7) gh = curr @ Whh^T + bhh                  (8192 x 768 x 256)
    wmma_gemm16<false, true, false><<<dim3(6, 64, 1), blk, 0, stream>>>(
        hb + H_HID, hb + H_WHH, ws + OFF_GH, gru_bhh, nullptr, 0,
        CB * CA, 3 * CR, CR, CR, CR, 3 * CR, 0, 0, 0);

    // 8) GRU gates -> new hidden (d_out) + f16 mirror
    {
        const int total = CB * CA * CR;
        gru_gates<<<dim3(total / 256), blk, 0, stream>>>(ws + OFF_GI, ws + OFF_GH, hidden,
                                                         o_hid, hb + H_NR, total);
    }

    // 9) output_active = new_regs @ out_w^T + b   (8192 x 512 x 256)
    wmma_gemm16<false, true, false><<<dim3(4, 64, 1), blk, 0, stream>>>(
        hb + H_NR, hb + H_OUTW, ws + OFF_OUTA, out_b, hb + H_OUTA, 0,
        CB * CA, CIN, CR, CR, CR, CIN, 0, 0, 0);

    // 10) proj = output_active @ agg_proj_w^T + b (8192 x 1024 x 512)
    wmma_gemm16<false, true, false><<<dim3(8, 64, 1), blk, 0, stream>>>(
        hb + H_OUTA, hb + H_PROJW, ws + OFF_PROJ, agg_pb, nullptr, 0,
        CB * CA, CH, CIN, CIN, CIN, CH, 0, 0, 0);

    // 11) w_attn logits = w_key @ mem^T           (per-b 256 x 2048 x 256)
    wmma_gemm16<false, true, false><<<dim3(16, 2, CB), blk, 0, stream>>>(
        hb + H_MMU + CD, hb + H_MEM, ws + OFF_WATT, nullptr, nullptr, 0,
        CA, CM, CD, CMW, CD, CM,
        (long long)CA * CMW, (long long)CM * CD, (long long)CA * CM);

    // 12) softmax * sigmoid(w_gate) (+ f16 mirror); gate column at mmu_out + 3D
    row_softmax<<<dim3(CB * CA), blk, 0, stream>>>(ws + OFF_WATT, hb + H_WATTG,
                                                   CM, 0.0625f,
                                                   ws + OFF_MMU + 3 * CD, CMW);

    // 13) total_w = clip(col-sum, 0, 1)
    colsum_clip<<<dim3(CM / 256, CB), blk, 0, stream>>>(ws + OFF_WATT, ws + OFF_TW);

    // 14) combined = gated_wattn^T @ w_val        (per-b 2048 x 256 x 256), reuse attn buffer
    wmma_gemm16<true, false, false><<<dim3(2, 16, CB), blk, 0, stream>>>(
        hb + H_WATTG, hb + H_MMU + 2 * CD, ws + OFF_ATTN, nullptr, nullptr, 0,
        CM, CD, CA, CM, CMW, CD,
        (long long)CA * CM, (long long)CA * CMW, (long long)CM * CD);

    // 15) ssd gate reduction -> agg_key, read_mask
    ssd_reduce<<<dim3(CB), blk, 0, stream>>>(ws + OFF_MMU, ws + OFF_AGGK, ws + OFF_MASK);

    // 16) ssd attention -> blended slot0
    ssd_slot0<<<dim3(CB), blk, 0, stream>>>(ws + OFF_AGGK, ssd_keys, ssd_vals,
                                            ws + OFF_MASK, memstate, ws + OFF_SLOT);

    // 17) final memory blend -> d_out
    {
        const int total = CB * CM * CD;
        mem_final<<<dim3(total / 256), blk, 0, stream>>>(memstate, ws + OFF_SLOT, ws + OFF_TW,
                                                         ws + OFF_ATTN, o_mem, total);
    }

    // 18) aggregation attention + LayerNorm -> d_out
    agg_ln<<<dim3(CB), blk, 0, stream>>>(ws + OFF_PROJ, agg_q, ln_g, ln_b, o_agg);
}
